// MultiheadAttention_88321707475479
// MI455X (gfx1250) — matrix operand order repaired and checked by reading, not compiled, not run on hardware
//
#include <hip/hip_runtime.h>
#include <hip/hip_bf16.h>

typedef unsigned short u16t;
typedef unsigned int   u32;
typedef __bf16 bf16_t;
typedef bf16_t v16bf __attribute__((ext_vector_type(16)));
typedef float  v8f   __attribute__((ext_vector_type(8)));
typedef u32    u32x4 __attribute__((ext_vector_type(4)));
typedef float  f32x4 __attribute__((ext_vector_type(4)));

#define E_DIM 1024
#define SEQ_L 2048
#define NB    4
#define NH    16
#define M_TOT 8192   // NB * SEQ_L

// ---------- helpers ----------
__device__ inline u16t f2bf(float f) {            // f32 -> bf16 RNE
  u32 u = __builtin_bit_cast(u32, f);
  u32 r = u + 0x7fffu + ((u >> 16) & 1u);
  return (u16t)(r >> 16);
}

union FragU {
  struct { u32x4 lo, hi; } q;
  v16bf v;
};

__device__ inline v8f wmma_bf16(v16bf a, v16bf b, v8f c) {
  return __builtin_amdgcn_wmma_f32_16x16x32_bf16(false, a, false, b, (short)0, c,
                                                 false, false);
}

// A fragment (16x32 bf16, M x K) from LDS stored [m][k] with given pitch (ushorts).
// Lane layout per ISA 7.12.2: lanes 0-15 row m, e0..7=K0..7, e8..15=K16..23;
// lanes 16-31 row m, e0..7=K8..15, e8..15=K24..31.
__device__ inline v16bf frag_a_lds(const u16t* s, int rowBase, int kBase, int lane, int pitch) {
  const u16t* p = s + (size_t)(rowBase + (lane & 15)) * pitch + kBase + ((lane >> 4) << 3);
  FragU f;
  f.q.lo = *(const u32x4*)(p);
  f.q.hi = *(const u32x4*)(p + 16);
  return f.v;
}

// B fragment (32x16 bf16, K x N) from LDS stored transposed [n][k].
// lanes 0-15: col n, e0..7=K0..7, e8..15=K16..23; lanes 16-31: col n, e0..7=K8..15, e8..15=K24..31.
__device__ inline v16bf frag_b_lds(const u16t* s, int rowBase, int kBase, int lane, int pitch) {
  const u16t* p = s + (size_t)(rowBase + (lane & 15)) * pitch + kBase + ((lane >> 4) << 3);
  FragU f;
  f.q.lo = *(const u32x4*)(p);
  f.q.hi = *(const u32x4*)(p + 16);
  return f.v;
}

// A fragment straight from global row-major [m][k] (16 consecutive rows).
__device__ inline v16bf frag_a_glb(const u16t* g, int stride, int lane, int kOff) {
  const u16t* p = g + (size_t)(lane & 15) * stride + kOff + ((lane >> 4) << 3);
  FragU f;
  f.q.lo = *(const u32x4*)(p);
  f.q.hi = *(const u32x4*)(p + 16);
  return f.v;
}

// ---------- kernel 1: x = seq + positional; emit f32 + bf16 ----------
__global__ __launch_bounds__(256) void prep_k(const float* __restrict__ seq,
                                              float* __restrict__ x32,
                                              u16t* __restrict__ xb) {
  size_t i = (size_t)blockIdx.x * 256 + threadIdx.x;
  int c = (int)(i & (E_DIM - 1));
  int l = (int)((i >> 10) & (SEQ_L - 1));
  const float kln = 9.210340371976184f / 512.0f;   // ln(10000)/512
  float pe;
  if (c < 512) pe = __sinf((float)l * __expf(-(float)c * kln));
  else         pe = __cosf((float)l * __expf(-(float)(c - 512) * kln));
  float v = seq[i] + pe;
  x32[i] = v;
  xb[i]  = f2bf(v);
}

// ---------- kernel 2a: f32 -> bf16 convert (keeps layout; used for lin_W) ----
__global__ __launch_bounds__(256) void cvt_k(const float* __restrict__ s,
                                             u16t* __restrict__ d) {
  size_t i = (size_t)blockIdx.x * 256 + threadIdx.x;
  d[i] = f2bf(s[i]);
}

// ---------- kernel 2b: f32 [K,N] -> bf16 [N,K] transpose-convert ----------
// One-time: hoists the GEMM B-transpose out of the GEMM hot loop.
__global__ __launch_bounds__(256) void cvt_t_k(const float* __restrict__ s,
                                               u16t* __restrict__ d,
                                               int N, int K) {
  __shared__ u16t t[32][33];
  const int n0 = blockIdx.x * 32;
  const int k0 = blockIdx.y * 32;
  const int tx = threadIdx.x & 31;       // blockDim = 256 -> 32x8 logical
  const int ty = threadIdx.x >> 5;       // 0..7
#pragma unroll
  for (int i = ty; i < 32; i += 8)
    t[i][tx] = f2bf(s[(size_t)(k0 + i) * N + n0 + tx]);   // t[k][n]
  __syncthreads();
#pragma unroll
  for (int i = ty; i < 32; i += 8)
    d[(size_t)(n0 + i) * K + k0 + tx] = t[tx][i];          // d[n][k] = s[k][n]
}

// ---------- kernel 3: tiled bf16 GEMM, f32 accum ----------
// C[M,N] = A[M,K] * Bt^T where Bt is [N,K] row-major (pre-transposed weights).
// Block tile 256x128, BK=64; 8 waves in 4x2: each wave 64x64 (4x4 wmma tiles)
// -> 1.0 ds_load_b128 per WMMA and 16 acc-tile register reuse.
// Register-staged double buffer: next tile's global loads issued before the
// current tile's 32 WMMAs, hiding HBM latency.
template <int OUTB, int OUTF, int BIAS, int RELU>
__global__ __launch_bounds__(256) void gemm_k(const u16t* __restrict__ A,
                                              const u16t* __restrict__ Bt,
                                              u16t* __restrict__ Cb,
                                              float* __restrict__ Cf,
                                              const float* __restrict__ bias,
                                              int M, int N, int K) {
  __shared__ __align__(16) u16t sA[256 * 72];   // [m][k], pitch 72 ushorts
  __shared__ __align__(16) u16t sB[128 * 72];   // [n][k]
  const int tid = threadIdx.x;
  const int lane = tid & 31;
  const int wave = tid >> 5;
  const int wm = wave >> 1, wn = wave & 1;      // 4x2 wave grid
  const int m0 = blockIdx.y * 256;
  const int n0 = blockIdx.x * 128;

  const int rowA = tid;                  // 0..255: full 64-elem k row of A
  const int rowB = tid >> 1;             // 0..127
  const int khB  = (tid & 1) * 32;       // 0 or 32
  const u16t* ga = A  + (size_t)(m0 + rowA) * K;
  const u16t* gb = Bt + (size_t)(n0 + rowB) * K + khB;
  u32x4* dA = (u32x4*)(sA + rowA * 72);
  u32x4* dB = (u32x4*)(sB + rowB * 72 + khB);

  v8f acc[4][4];
#pragma unroll
  for (int mt = 0; mt < 4; ++mt)
#pragma unroll
    for (int nt = 0; nt < 4; ++nt) acc[mt][nt] = v8f{};

  // prologue: tile 0 -> regs -> LDS
  u32x4 ra[8], rb[4];
#pragma unroll
  for (int i = 0; i < 8; ++i) ra[i] = *(const u32x4*)(ga + i * 8);
#pragma unroll
  for (int i = 0; i < 4; ++i) rb[i] = *(const u32x4*)(gb + i * 8);
#pragma unroll
  for (int i = 0; i < 8; ++i) dA[i] = ra[i];
#pragma unroll
  for (int i = 0; i < 4; ++i) dB[i] = rb[i];

  for (int k0 = 0; k0 < K; k0 += 64) {
    __syncthreads();                        // LDS tile k0 ready
    const bool more = (k0 + 64) < K;
    if (more) {                             // issue next tile's loads early
      const u16t* na = ga + k0 + 64;
      const u16t* nb = gb + k0 + 64;
#pragma unroll
      for (int i = 0; i < 8; ++i) ra[i] = *(const u32x4*)(na + i * 8);
#pragma unroll
      for (int i = 0; i < 4; ++i) rb[i] = *(const u32x4*)(nb + i * 8);
    }
#pragma unroll
    for (int ks = 0; ks < 2; ++ks) {
      v16bf af[4], bfr[4];
#pragma unroll
      for (int mt = 0; mt < 4; ++mt)
        af[mt] = frag_a_lds(sA, wm * 64 + mt * 16, ks * 32, lane, 72);
#pragma unroll
      for (int nt = 0; nt < 4; ++nt)
        bfr[nt] = frag_b_lds(sB, wn * 64 + nt * 16, ks * 32, lane, 72);
#pragma unroll
      for (int mt = 0; mt < 4; ++mt)
#pragma unroll
        for (int nt = 0; nt < 4; ++nt)
          acc[mt][nt] = wmma_bf16(af[mt], bfr[nt], acc[mt][nt]);
    }
    __syncthreads();                        // compute done before overwrite
    if (more) {
#pragma unroll
      for (int i = 0; i < 8; ++i) dA[i] = ra[i];
#pragma unroll
      for (int i = 0; i < 4; ++i) dB[i] = rb[i];
    }
  }

  // epilogue: C layout — col = lane&15, rows j + 8*(lane>=16)
#pragma unroll
  for (int mt = 0; mt < 4; ++mt) {
#pragma unroll
    for (int nt = 0; nt < 4; ++nt) {
      int col   = n0 + wn * 64 + nt * 16 + (lane & 15);
      int rbase = m0 + wm * 64 + mt * 16 + ((lane >> 4) << 3);
      float bv = BIAS ? bias[col] : 0.0f;
#pragma unroll
      for (int j = 0; j < 8; ++j) {
        float v = acc[mt][nt][j] + bv;
        if (RELU) v = v > 0.0f ? v : 0.0f;
        size_t idx = (size_t)(rbase + j) * N + col;
        if (OUTF) Cf[idx] = v;
        if (OUTB) Cb[idx] = f2bf(v);
      }
    }
  }
}

// ---------- kernel 4: kv[b,h] = (K^T V) / 8, 64x64 per (b,h) ----------
// One block per (b,h); all 8 waves share the K-chunk; each wave owns 2 of the
// 16 16x16 output tiles, so no cross-wave reduction is needed.
__global__ __launch_bounds__(256) void kv_k(const u16t* __restrict__ kq,
                                            const u16t* __restrict__ vq,
                                            u16t* __restrict__ kvp) {
  __shared__ __align__(16) u16t sA[64 * 40];   // [d1][l]
  __shared__ __align__(16) u16t sB[64 * 40];   // [d2][l]
  const int bh = blockIdx.x;
  const int b = bh >> 4, h = bh & 15;
  const int hb = h * 64;
  const int tid = threadIdx.x, lane = tid & 31, wave = tid >> 5;
  const int t0 = wave * 2;
  const int mt0 = t0 >> 2,       nt0 = t0 & 3;
  const int mt1 = (t0 + 1) >> 2, nt1 = (t0 + 1) & 3;
  v8f acc0 = v8f{}, acc1 = v8f{};

  const size_t rowB = (size_t)b * SEQ_L;
  const int ll = tid >> 3;            // 0..31 (l within chunk)
  const int ds = (tid & 7) * 8;       // 0..56 (d offset)

  for (int c = 0; c < SEQ_L / 32; ++c) {
    const size_t r = (rowB + (size_t)c * 32 + ll) * E_DIM + hb + ds;
    u16t tk[8] __attribute__((aligned(16)));
    u16t tv[8] __attribute__((aligned(16)));
    *(u32x4*)tk = *(const u32x4*)(kq + r);
    *(u32x4*)tv = *(const u32x4*)(vq + r);
#pragma unroll
    for (int i = 0; i < 8; ++i) {
      sA[(ds + i) * 40 + ll] = tk[i];   // transpose: [d1][l]
      sB[(ds + i) * 40 + ll] = tv[i];   // transpose: [d2][l]
    }
    __syncthreads();
    v16bf a0 = frag_a_lds(sA, mt0 * 16, 0, lane, 40);
    v16bf b0 = frag_b_lds(sB, nt0 * 16, 0, lane, 40);
    acc0 = wmma_bf16(a0, b0, acc0);
    v16bf a1 = frag_a_lds(sA, mt1 * 16, 0, lane, 40);
    v16bf b1 = frag_b_lds(sB, nt1 * 16, 0, lane, 40);
    acc1 = wmma_bf16(a1, b1, acc1);
    __syncthreads();
  }

  u16t* out = kvp + (size_t)bh * 4096;
  {
    int col = nt0 * 16 + (lane & 15);
    int rb  = mt0 * 16 + ((lane >> 4) << 3);
#pragma unroll
    for (int j = 0; j < 8; ++j) out[(rb + j) * 64 + col] = f2bf(acc0[j] * 0.125f);
  }
  {
    int col = nt1 * 16 + (lane & 15);
    int rb  = mt1 * 16 + ((lane >> 4) << 3);
#pragma unroll
    for (int j = 0; j < 8; ++j) out[(rb + j) * 64 + col] = f2bf(acc1[j] * 0.125f);
  }
}

// ---------- kernel 5: ctx = Q @ kv per head ----------
// grid.x = (b,h), grid.y = 128-row tile. A frags straight from global q.
__global__ __launch_bounds__(256) void ctx_k(const u16t* __restrict__ qm,
                                             const u16t* __restrict__ kvp,
                                             u16t* __restrict__ ctx) {
  __shared__ __align__(16) u16t sB[64 * 72];   // [d2][d1], pitch 72
  const int bh = blockIdx.x;
  const int b = bh >> 4, h = bh & 15;
  const int hb = h * 64;
  const int tid = threadIdx.x, lane = tid & 31, wave = tid >> 5;

  {  // load whole 64x64 kv, transpose into [n=d2][k=d1]
    int d1  = tid >> 2;           // 0..63
    int d2s = (tid & 3) * 16;     // 0,16,32,48
    const u16t* src = kvp + (size_t)bh * 4096 + d1 * 64 + d2s;
    u16t tmp[16] __attribute__((aligned(16)));
    *(u32x4*)(tmp)     = *(const u32x4*)(src);
    *(u32x4*)(tmp + 8) = *(const u32x4*)(src + 8);
#pragma unroll
    for (int i = 0; i < 16; ++i) sB[(d2s + i) * 72 + d1] = tmp[i];
  }
  __syncthreads();

  const size_t r0 = (size_t)b * SEQ_L + (size_t)blockIdx.y * 128 + wave * 16;
  const u16t* qbase = qm + r0 * E_DIM + hb;
  v8f acc[4];
#pragma unroll
  for (int nt = 0; nt < 4; ++nt) acc[nt] = v8f{};

#pragma unroll
  for (int kc = 0; kc < 2; ++kc) {
    v16bf a = frag_a_glb(qbase, E_DIM, lane, kc * 32);
#pragma unroll
    for (int nt = 0; nt < 4; ++nt) {
      v16bf bb = frag_b_lds(sB, nt * 16, kc * 32, lane, 72);
      acc[nt] = wmma_bf16(a, bb, acc[nt]);
    }
  }

#pragma unroll
  for (int nt = 0; nt < 4; ++nt) {
    int col = hb + nt * 16 + (lane & 15);
    size_t rb = r0 + ((lane >> 4) << 3);
#pragma unroll
    for (int j = 0; j < 8; ++j) ctx[(rb + j) * E_DIM + col] = f2bf(acc[nt][j]);
  }
}

// ---------- kernel 6: fused residual + LayerNorm ----------
template <int WB16>
__global__ __launch_bounds__(256) void ln_k(const float* __restrict__ X,
                                            const float* __restrict__ R,
                                            const float* __restrict__ g,
                                            const float* __restrict__ bta,
                                            float* __restrict__ o32,
                                            u16t* __restrict__ ob) {
  const int r = blockIdx.x;
  const int tid = threadIdx.x;
  const size_t base = (size_t)r * E_DIM + tid * 4;
  f32x4 x  = *(const f32x4*)(X + base);
  f32x4 rr = *(const f32x4*)(R + base);
  f32x4 v = x + rr;
  float s  = v[0] + v[1] + v[2] + v[3];
  float qq = v[0] * v[0] + v[1] * v[1] + v[2] * v[2] + v[3] * v[3];
#pragma unroll
  for (int o = 16; o > 0; o >>= 1) {
    s  += __shfl_xor(s, o, 32);
    qq += __shfl_xor(qq, o, 32);
  }
  __shared__ float ss[8], sq[8];
  const int lane = tid & 31, wave = tid >> 5;
  if (lane == 0) { ss[wave] = s; sq[wave] = qq; }
  __syncthreads();
  float S = 0.0f, Q = 0.0f;
#pragma unroll
  for (int i = 0; i < 8; ++i) { S += ss[i]; Q += sq[i]; }
  const float mean = S * (1.0f / E_DIM);
  const float var  = Q * (1.0f / E_DIM) - mean * mean;
  const float inv  = rsqrtf(var + 1e-5f);
#pragma unroll
  for (int j = 0; j < 4; ++j) {
    int c = tid * 4 + j;
    float y = (v[j] - mean) * inv * g[c] + bta[c];
    o32[base + j] = y;
    if (WB16) ob[base + j] = f2bf(y);
  }
}

// ---------- launch ----------
extern "C" void kernel_launch(void* const* d_in, const int* in_sizes, int n_in,
                              void* d_out, int out_size, void* d_ws, size_t ws_size,
                              hipStream_t stream) {
  (void)in_sizes; (void)n_in; (void)out_size; (void)ws_size;
  const float* seq  = (const float*)d_in[0];
  const float* Wq   = (const float*)d_in[1];
  const float* Wk   = (const float*)d_in[2];
  const float* Wv   = (const float*)d_in[3];
  const float* Wo   = (const float*)d_in[4];
  const float* ln1g = (const float*)d_in[5];
  const float* ln1b = (const float*)d_in[6];
  const float* linW = (const float*)d_in[7];   // [out,in] = [N,K] already
  const float* linb = (const float*)d_in[8];
  const float* ln2g = (const float*)d_in[9];
  const float* ln2b = (const float*)d_in[10];
  float* out = (float*)d_out;

  char* w = (char*)d_ws;
  const size_t MB = 1ull << 20;
  float* x32 = (float*)(w + 0);          // 32MB: x, later h (in-place)
  u16t*  xb  = (u16t*)(w + 32 * MB);     // 16MB: x bf16, later h bf16
  u16t*  wqb = (u16t*)(w + 48 * MB);     // 2MB each, all stored [N,K]
  u16t*  wkb = (u16t*)(w + 50 * MB);
  u16t*  wvb = (u16t*)(w + 52 * MB);
  u16t*  wob = (u16t*)(w + 54 * MB);
  u16t*  wlb = (u16t*)(w + 56 * MB);
  u16t*  qb  = (u16t*)(w + 58 * MB);     // 16MB
  u16t*  kb  = (u16t*)(w + 74 * MB);     // 16MB: k, later ctx
  u16t*  vb  = (u16t*)(w + 90 * MB);     // 16MB
  u16t*  kvb = (u16t*)(w + 106 * MB);    // 512KB
  float* af  = (float*)(w + 107 * MB);   // 32MB: attn_out, later ff

  // 1. positional + split precision
  prep_k<<<(M_TOT * E_DIM) / 256, 256, 0, stream>>>(seq, x32, xb);
  // 2. weights -> bf16; projection weights transposed to [N,K] once, so the
  //    GEMM hot loop has zero LDS transpose work. lin_W is already [N,K].
  dim3 tg(E_DIM / 32, E_DIM / 32);
  cvt_t_k<<<tg, 256, 0, stream>>>(Wq, wqb, E_DIM, E_DIM);
  cvt_t_k<<<tg, 256, 0, stream>>>(Wk, wkb, E_DIM, E_DIM);
  cvt_t_k<<<tg, 256, 0, stream>>>(Wv, wvb, E_DIM, E_DIM);
  cvt_t_k<<<tg, 256, 0, stream>>>(Wo, wob, E_DIM, E_DIM);
  cvt_k<<<(E_DIM * E_DIM) / 256, 256, 0, stream>>>(linW, wlb);

  dim3 gg(E_DIM / 128, M_TOT / 256);
  // 3. q,k,v projections (bf16 out)
  gemm_k<1, 0, 0, 0><<<gg, 256, 0, stream>>>(xb, wqb, qb, nullptr, nullptr, M_TOT, E_DIM, E_DIM);
  gemm_k<1, 0, 0, 0><<<gg, 256, 0, stream>>>(xb, wkb, kb, nullptr, nullptr, M_TOT, E_DIM, E_DIM);
  gemm_k<1, 0, 0, 0><<<gg, 256, 0, stream>>>(xb, wvb, vb, nullptr, nullptr, M_TOT, E_DIM, E_DIM);
  // 4. kv = K^T V / 8 per head
  kv_k<<<NB * NH, 256, 0, stream>>>(kb, vb, kvb);
  // 5. ctx = Q @ kv  (writes into kb; k is dead)
  ctx_k<<<dim3(NB * NH, SEQ_L / 128), 256, 0, stream>>>(qb, kvb, kb);
  // 6. attn_out = ctx @ Wo (f32 out)
  gemm_k<0, 1, 0, 0><<<gg, 256, 0, stream>>>(kb, wob, nullptr, af, nullptr, M_TOT, E_DIM, E_DIM);
  // 7. h = LN(x + attn_out)  (in-place over x32 / xb)
  ln_k<1><<<M_TOT, 256, 0, stream>>>(x32, af, ln1g, ln1b, x32, xb);
  // 8. ff = relu(h @ linW^T + b)  (f32 out, overwrites attn buffer)
  gemm_k<0, 1, 1, 1><<<gg, 256, 0, stream>>>(xb, wlb, nullptr, af, linb, M_TOT, E_DIM, E_DIM);
  // 9. out = LN(ff + h)
  ln_k<0><<<M_TOT, 256, 0, stream>>>(af, x32, ln2g, ln2b, out, nullptr);
}
